// LocalSmoother_30434138259738
// MI455X (gfx1250) — matrix operand order repaired and checked by reading, not compiled, not run on hardware
//
#include <hip/hip_runtime.h>
#include <hip/hip_bf16.h>

typedef __attribute__((ext_vector_type(16))) _Float16 v16h;
typedef __attribute__((ext_vector_type(8)))  _Float16 v8h;
typedef __attribute__((ext_vector_type(4)))  _Float16 v4h;
typedef __attribute__((ext_vector_type(8)))  float    v8f;

#define D_MODEL 1024
#define N_HEADS 16
#define D_HEAD  64
#define CHUNK   32
#define SEQ_B   4
#define SEQ_L   4096
#define ROWS    (SEQ_B * SEQ_L)          // 16384
#define NCHUNK  (SEQ_L / CHUNK)          // 128

// ---------------------------------------------------------------- utilities

__global__ void cvt_f32_to_f16_kernel(const float* __restrict__ src,
                                      _Float16* __restrict__ dst, int nvec) {
    int i = blockIdx.x * blockDim.x + threadIdx.x;
    if (i < nvec) {
        float4 v = ((const float4*)src)[i];
        v4h o;
        o[0] = (_Float16)v.x; o[1] = (_Float16)v.y;
        o[2] = (_Float16)v.z; o[3] = (_Float16)v.w;
        ((v4h*)dst)[i] = o;
    }
}

__global__ void rope_tables_kernel(float* __restrict__ ct, float* __restrict__ st) {
    int i = blockIdx.x * blockDim.x + threadIdx.x;   // 0..2047 = pos*64 + d
    if (i < CHUNK * D_HEAD) {
        int pos = i >> 6;
        int d   = i & 63;
        float inv_freq = powf(10000.0f, -((float)(2 * (d & 31))) / 64.0f);
        float f = (float)pos * inv_freq;
        ct[i] = cosf(f);
        st[i] = sinf(f);
    }
}

// ------------------------------------------------------------- LayerNorm

__global__ void ln_kernel(const float* __restrict__ x,
                          const float* __restrict__ gamma,
                          const float* __restrict__ beta,
                          _Float16* __restrict__ out) {
    __shared__ float rs[256], rq[256];
    const int row = blockIdx.x;
    const int tid = threadIdx.x;
    const float4 v = *(const float4*)(x + (size_t)row * D_MODEL + tid * 4);
    float s = v.x + v.y + v.z + v.w;
    float q = v.x * v.x + v.y * v.y + v.z * v.z + v.w * v.w;
    rs[tid] = s; rq[tid] = q;
    __syncthreads();
    for (int off = 128; off > 0; off >>= 1) {
        if (tid < off) { rs[tid] += rs[tid + off]; rq[tid] += rq[tid + off]; }
        __syncthreads();
    }
    const float mean = rs[0] * (1.0f / D_MODEL);
    const float var  = rq[0] * (1.0f / D_MODEL) - mean * mean;
    const float inv  = rsqrtf(var + 1e-5f);
    const int c = tid * 4;
    v4h o;
    o[0] = (_Float16)((v.x - mean) * inv * gamma[c + 0] + beta[c + 0]);
    o[1] = (_Float16)((v.y - mean) * inv * gamma[c + 1] + beta[c + 1]);
    o[2] = (_Float16)((v.z - mean) * inv * gamma[c + 2] + beta[c + 2]);
    o[3] = (_Float16)((v.w - mean) * inv * gamma[c + 3] + beta[c + 3]);
    *(v4h*)(out + (size_t)row * D_MODEL + c) = o;
}

// --------------------------------------------------- WMMA GEMM  C = A * W^T
// A: (M,K) f16 row-major.  W: (N,K) f16 row-major (so B[k][n] = W[n][k]).
// Each wave computes a 32x64 tile of C: 2 A-frags x 4 B-frags, K-step 32.
// If Cf16 != nullptr: store f16. Else: Cf32[row,col] = residual + acc (f32).

__global__ void wmma_gemm_kernel(const _Float16* __restrict__ A,
                                 const _Float16* __restrict__ W,
                                 _Float16* __restrict__ Cf16,
                                 float* __restrict__ Cf32,
                                 const float* __restrict__ residual,
                                 int M, int N, int K) {
    const int lane = threadIdx.x & 31;
    const int wave = threadIdx.x >> 5;
    const int wg   = blockIdx.x * (blockDim.x >> 5) + wave;
    const int nt64 = N >> 6;
    const int mtile = wg / nt64;
    const int ntile = wg % nt64;
    if (mtile >= (M >> 5)) return;                    // wave-uniform guard
    const int m0 = mtile << 5;
    const int n0 = ntile << 6;

    const int lm   = lane & 15;
    const int koff = (lane < 16) ? 0 : 8;             // A-frag lane-half K offset
    const int kb   = (lane < 16) ? 0 : 16;            // W-prefetch lane-half K offset

    const _Float16* aptr[2];
    const _Float16* wptr[4];
#pragma unroll
    for (int am = 0; am < 2; ++am)
        aptr[am] = A + (size_t)(m0 + am * 16 + lm) * K;
#pragma unroll
    for (int bn = 0; bn < 4; ++bn)
        wptr[bn] = W + (size_t)(n0 + bn * 16 + lm) * K;

    v8f acc[2][4];
#pragma unroll
    for (int am = 0; am < 2; ++am)
#pragma unroll
        for (int bn = 0; bn < 4; ++bn)
            acc[am][bn] = (v8f)(0.0f);

    for (int k0 = 0; k0 < K; k0 += 32) {
        // prefetch next K tile into WGP-level caches (global_prefetch_b8)
        if (k0 + 32 < K) {
            __builtin_prefetch(aptr[0] + k0 + 32 + koff, 0, 3);
            __builtin_prefetch(wptr[0] + k0 + 32 + kb, 0, 3);
        }
        v16h afrag[2];
#pragma unroll
        for (int am = 0; am < 2; ++am) {
            v8h lo = *(const v8h*)(aptr[am] + k0 + koff);        // K = koff..koff+7
            v8h hi = *(const v8h*)(aptr[am] + k0 + 16 + koff);   // K = 16+koff..23+koff
#pragma unroll
            for (int i = 0; i < 8; ++i) { afrag[am][i] = lo[i]; afrag[am][i + 8] = hi[i]; }
        }
        v16h bfrag[4];
#pragma unroll
        for (int bn = 0; bn < 4; ++bn) {
            v8h lo = *(const v8h*)(wptr[bn] + k0 + koff);        // K = koff..koff+7
            v8h hi = *(const v8h*)(wptr[bn] + k0 + 16 + koff);   // K = 16+koff..23+koff
#pragma unroll
            for (int i = 0; i < 8; ++i) { bfrag[bn][i] = lo[i]; bfrag[bn][i + 8] = hi[i]; }
        }
#pragma unroll
        for (int am = 0; am < 2; ++am)
#pragma unroll
            for (int bn = 0; bn < 4; ++bn)
                acc[am][bn] = __builtin_amdgcn_wmma_f32_16x16x32_f16(
                    false, afrag[am], false, bfrag[bn],
                    (short)0, acc[am][bn], false, false);
    }

    const int rbase = (lane < 16) ? 0 : 8;            // C/D: vgpr r -> row r / r+8
#pragma unroll
    for (int am = 0; am < 2; ++am) {
#pragma unroll
        for (int bn = 0; bn < 4; ++bn) {
            const size_t col = (size_t)(n0 + bn * 16 + lm);
#pragma unroll
            for (int r = 0; r < 8; ++r) {
                const size_t row = (size_t)(m0 + am * 16 + rbase + r);
                if (Cf16) {
                    Cf16[row * N + col] = (_Float16)acc[am][bn][r];
                } else {
                    Cf32[row * N + col] = residual[row * N + col] + acc[am][bn][r];
                }
            }
        }
    }
}

// ------------------------------------------------------------- attention

__device__ __forceinline__ void rope8(const _Float16* __restrict__ rowp, int d0,
                                      const float* __restrict__ c8,
                                      const float* __restrict__ s8,
                                      _Float16* __restrict__ out) {
    v8h xv = *(const v8h*)(rowp + d0);
    v8h yv = *(const v8h*)(rowp + (d0 ^ 32));
    const float sgn = (d0 & 32) ? 1.0f : -1.0f;
#pragma unroll
    for (int i = 0; i < 8; ++i)
        out[i] = (_Float16)((float)xv[i] * c8[i] + sgn * (float)yv[i] * s8[i]);
}

__global__ void attn_kernel(const _Float16* __restrict__ qkv,
                            const float* __restrict__ mask,
                            const float* __restrict__ ct,
                            const float* __restrict__ st,
                            _Float16* __restrict__ ao) {
    __shared__ float    S[4][CHUNK][CHUNK + 1];   // scores / probs
    __shared__ _Float16 Vs[4][CHUNK * D_HEAD];    // async-staged V chunk per wave
    const int tid  = threadIdx.x;
    const int lane = tid & 31;
    const int w    = tid >> 5;
    const int id   = blockIdx.x * 4 + w;              // chunk-head id, exact grid
    const int b    = id >> 11;                        // / (16*128)
    const int h    = (id >> 7) & 15;
    const int n    = id & 127;
    const size_t rowbase = ((size_t)b << 12) + ((size_t)n << 5);  // token row base
    const int lm   = lane & 15;
    const int koff = (lane < 16) ? 0 : 8;
    const int kb   = (lane < 16) ? 0 : 16;
    const size_t hoff = (size_t)h * D_HEAD;

    // ---- kick off async DMA of this wave's V chunk (32 x 64 f16) into LDS.
    // Each lane copies one V row (128 B) as 8 x b128 async transfers; the DMA
    // overlaps the RoPE + QK^T + softmax work below (tracked by ASYNCcnt).
    {
        const _Float16* vrow = qkv + (rowbase + lane) * (3 * D_MODEL) + 2 * D_MODEL + hoff;
        unsigned int lds_base = (unsigned int)(uintptr_t)(&Vs[w][0]) + (unsigned int)lane * 128u;
#pragma unroll
        for (int i = 0; i < 8; ++i) {
            const void* g = (const void*)(vrow + i * 8);
            unsigned int l = lds_base + (unsigned int)i * 16u;
            asm volatile("global_load_async_to_lds_b128 %0, %1, off"
                         :: "v"(l), "v"(g) : "memory");
        }
    }

    // ---- Q A-fragments with RoPE: aq[mt][ks]  (rows m = mt*16+lm, pos = m)
    v16h aq[2][2];
#pragma unroll
    for (int mt = 0; mt < 2; ++mt) {
        const int m = mt * 16 + lm;
        const _Float16* qrow = qkv + (rowbase + m) * (3 * D_MODEL) + hoff;
        const float* cbase = ct + m * D_HEAD;
        const float* sbase = st + m * D_HEAD;
#pragma unroll
        for (int ks = 0; ks < 2; ++ks) {
            _Float16 t[16];
            const int d0 = ks * 32 + koff;
            const int d1 = ks * 32 + 16 + koff;
            rope8(qrow, d0, cbase + d0, sbase + d0, t);
            rope8(qrow, d1, cbase + d1, sbase + d1, t + 8);
#pragma unroll
            for (int i = 0; i < 16; ++i) aq[mt][ks][i] = t[i];
        }
    }
    // ---- K B-fragments with RoPE: bk[nt][ks]  (col j = nt*16+lm, pos = j)
    v16h bk[2][2];
#pragma unroll
    for (int nt = 0; nt < 2; ++nt) {
        const int j = nt * 16 + lm;
        const _Float16* krow = qkv + (rowbase + j) * (3 * D_MODEL) + D_MODEL + hoff;
        const float* cbase = ct + j * D_HEAD;
        const float* sbase = st + j * D_HEAD;
#pragma unroll
        for (int ks = 0; ks < 2; ++ks) {
            _Float16 t[16];
            const int d0 = ks * 32 + koff;
            const int d1 = ks * 32 + 16 + koff;
            rope8(krow, d0, cbase + d0, sbase + d0, t);
            rope8(krow, d1, cbase + d1, sbase + d1, t + 8);
#pragma unroll
            for (int i = 0; i < 16; ++i) bk[nt][ks][i] = t[i];
        }
    }
    // ---- S = (Q Kt) * scale, masked -> LDS
#pragma unroll
    for (int mt = 0; mt < 2; ++mt) {
#pragma unroll
        for (int nt = 0; nt < 2; ++nt) {
            v8f acc = (v8f)(0.0f);
#pragma unroll
            for (int ks = 0; ks < 2; ++ks)
                acc = __builtin_amdgcn_wmma_f32_16x16x32_f16(
                    false, aq[mt][ks], false, bk[nt][ks], (short)0, acc, false, false);
            const int jj = nt * 16 + lm;
            const float mv = mask[(size_t)b * SEQ_L + n * CHUNK + jj];
            const int mrow = mt * 16 + ((lane < 16) ? 0 : 8);
#pragma unroll
            for (int r = 0; r < 8; ++r)
                S[w][mrow + r][jj] = (mv == 0.0f) ? -__builtin_inff() : acc[r] * 0.125f;
        }
    }
    __syncthreads();
    // ---- softmax: 128 threads, one (wave,row) each
    {
        const int ww = tid >> 5;
        const int m  = tid & 31;
        float mx = -__builtin_inff();
        for (int j = 0; j < CHUNK; ++j) mx = fmaxf(mx, S[ww][m][j]);
        if (mx == -__builtin_inff()) {
            for (int j = 0; j < CHUNK; ++j) S[ww][m][j] = 0.0f;   // nan_to_num path
        } else {
            float sum = 0.0f;
            for (int j = 0; j < CHUNK; ++j) sum += expf(S[ww][m][j] - mx);
            const float inv = 1.0f / sum;
            for (int j = 0; j < CHUNK; ++j) S[ww][m][j] = expf(S[ww][m][j] - mx) * inv;
        }
    }
    __syncthreads();
    // ---- P A-fragments (f16) from LDS
    v16h ap[2];
#pragma unroll
    for (int mt = 0; mt < 2; ++mt) {
        const int m = mt * 16 + lm;
#pragma unroll
        for (int i = 0; i < 8; ++i) {
            ap[mt][i]     = (_Float16)S[w][m][koff + i];
            ap[mt][i + 8] = (_Float16)S[w][m][16 + koff + i];
        }
    }
    // ---- wait for this wave's V DMA, then O = P * V from LDS
    asm volatile("s_wait_asynccnt 0x0" ::: "memory");
#pragma unroll
    for (int nt = 0; nt < 4; ++nt) {
        const int d = nt * 16 + lm;
        v16h bv;
#pragma unroll
        for (int i = 0; i < 16; ++i) {
            const int j = (i < 8) ? (koff + i) : (16 + koff + (i - 8));     // key position for this K index
            bv[i] = Vs[w][j * D_HEAD + d];
        }
#pragma unroll
        for (int mt = 0; mt < 2; ++mt) {
            v8f o = __builtin_amdgcn_wmma_f32_16x16x32_f16(
                false, ap[mt], false, bv, (short)0, (v8f)(0.0f), false, false);
            const int mrow = mt * 16 + ((lane < 16) ? 0 : 8);
#pragma unroll
            for (int r = 0; r < 8; ++r)
                ao[(rowbase + mrow + r) * D_MODEL + hoff + d] = (_Float16)o[r];
        }
    }
}

// ------------------------------------------------------------- launcher

extern "C" void kernel_launch(void* const* d_in, const int* in_sizes, int n_in,
                              void* d_out, int out_size, void* d_ws, size_t ws_size,
                              hipStream_t stream) {
    const float* x     = (const float*)d_in[0];
    const float* mask  = (const float*)d_in[1];
    const float* gamma = (const float*)d_in[2];
    const float* beta  = (const float*)d_in[3];
    const float* Wqkv  = (const float*)d_in[4];
    const float* Wout  = (const float*)d_in[5];
    float* out = (float*)d_out;

    char* ws = (char*)d_ws;
    _Float16* xn   = (_Float16*)(ws + 0);                 //  32 MB : ln output f16
    _Float16* wq16 = (_Float16*)(ws + 33554432ull);       //   6 MB : W_qkv f16
    _Float16* wo16 = (_Float16*)(ws + 39845888ull);       //   2 MB : W_out f16
    _Float16* qkv  = (_Float16*)(ws + 41943040ull);       //  96 MB : qkv f16
    _Float16* ao   = (_Float16*)(ws + 142606336ull);      //  32 MB : attn out f16
    float*    ct   = (float*)(ws + 176160768ull);         //  8 KB : rope cos
    float*    st   = (float*)(ws + 176168960ull);         //  8 KB : rope sin

    // weight conversion (f32 -> f16)
    cvt_f32_to_f16_kernel<<<3072, 256, 0, stream>>>(Wqkv, wq16, (3 * D_MODEL * D_MODEL) / 4);
    cvt_f32_to_f16_kernel<<<1024, 256, 0, stream>>>(Wout, wo16, (D_MODEL * D_MODEL) / 4);
    // rope tables
    rope_tables_kernel<<<8, 256, 0, stream>>>(ct, st);
    // layernorm
    ln_kernel<<<ROWS, 256, 0, stream>>>(x, gamma, beta, xn);
    // QKV GEMM: (16384 x 3072) = xn (16384x1024) * Wqkv^T ; waves = 512*48 = 24576
    wmma_gemm_kernel<<<6144, 128, 0, stream>>>(xn, wq16, qkv, nullptr, nullptr,
                                               ROWS, 3 * D_MODEL, D_MODEL);
    // local attention: 4*16*128 = 8192 chunk-heads, 4 per block
    attn_kernel<<<2048, 128, 0, stream>>>(qkv, mask, ct, st, ao);
    // output projection + residual: waves = 512*16 = 8192
    wmma_gemm_kernel<<<2048, 128, 0, stream>>>(ao, wo16, nullptr, out, x,
                                               ROWS, D_MODEL, D_MODEL);
}
